// LidarYOLOX_15599321219381
// MI455X (gfx1250) — matrix operand order repaired and checked by reading, not compiled, not run on hardware
//
#include <hip/hip_runtime.h>
#include <hip/hip_bf16.h>

// ---------------------------------------------------------------------------
// MI455X (gfx1250) implementation notes:
//  * All three edge-MLP layers run through V_WMMA_F32_16X16X32_BF16 (8x the
//    K-depth of the f32 WMMA path), f32 accumulation.
//  * Weights are re-packed each launch into per-lane B-fragment dword order
//    (cheap: <= 256x256 elements) so GEMM inner loops are two b128 loads/lane.
//  * segment_max -> monotonic-uint encoding + global_atomic_max_u32.
//  * scatter-add image (300MB) + 4x4 maxpool; image & max-buffers zeroed with
//    hipMemsetAsync on the capture stream (graph-safe, deterministic).
// ---------------------------------------------------------------------------

typedef __attribute__((ext_vector_type(16))) __bf16 v16bf;
typedef __attribute__((ext_vector_type(8)))  float  v8f;

union ABu { unsigned u[8]; v16bf v; };

static __device__ __forceinline__ unsigned short f2bf(float f) {
  union { float f; unsigned u; } x; x.f = f;
  unsigned r = x.u + 0x7FFFu + ((x.u >> 16) & 1u);   // round-to-nearest-even
  return (unsigned short)(r >> 16);
}

// order-preserving float->uint encoding (for atomic max)
static __device__ __forceinline__ unsigned encf(float f) {
  union { float f; unsigned u; } x; x.f = f;
  return (x.u & 0x80000000u) ? ~x.u : (x.u | 0x80000000u);
}
static __device__ __forceinline__ float decf(unsigned u) {
  union { unsigned u; float f; } x;
  x.u = (u & 0x80000000u) ? (u & 0x7FFFFFFFu) : ~u;
  return x.f;
}

// ---------------------------------------------------------------------------
// Weight pre-pack: f32 [K,N] row-major -> bf16-pair dwords in exact B-fragment
// order: bp[((ks*NT+nt)*256) + lane*8 + v] = {W[k][n], W[k+1][n]} with
// n = nt*16 + lane%16, k = ks*32 + 2v + 16*(lane/16). Zero-pad K >= Kreal.
// ---------------------------------------------------------------------------
__global__ void pack_w(const float* __restrict__ w, unsigned* __restrict__ bp,
                       int Kreal, int Kp, int Nout) {
  int i = blockIdx.x * 256 + threadIdx.x;
  int total = (Kp / 32) * (Nout / 16) * 256;
  if (i >= total) return;
  int v    = i & 7;
  int lane = (i >> 3) & 31;
  int tile = i >> 8;
  int NT   = Nout / 16;
  int nt   = tile % NT;
  int ks   = tile / NT;
  int hf   = lane >> 4;
  int n    = nt * 16 + (lane & 15);
  int k0   = ks * 32 + 2 * v + 16 * hf;
  unsigned lo = (k0     < Kreal) ? f2bf(w[(long)k0       * Nout + n]) : 0u;
  unsigned hi = (k0 + 1 < Kreal) ? f2bf(w[(long)(k0 + 1) * Nout + n]) : 0u;
  bp[i] = lo | (hi << 16);
}

// in-place decode: encoded-uint max buffer -> f32 features
__global__ void decode_k(unsigned* __restrict__ buf, long total) {
  long i = (long)blockIdx.x * 256 + threadIdx.x;
  if (i >= total) return;
  float f = decf(buf[i]);
  ((float*)buf)[i] = f;
}

// ---------------------------------------------------------------------------
// Fused PointNetConv layer. 32 edges / workgroup, 8 wave32.
//   msg = [feat[src], pos[src]-pos[dst]]  (bf16, LDS)
//   H1  = relu(msg @ W1 + b1)             (WMMA, bf16 in LDS)
//   H2  = H1 @ W2 + b2                    (WMMA)
//   enc[dst] = atomic-max(enc(H2))
// ---------------------------------------------------------------------------
template<int CF, int DOUT>
__global__ __launch_bounds__(256)
void layer_kernel(const float* __restrict__ feat, const float* __restrict__ pos,
                  const int* __restrict__ srcArr, const int* __restrict__ dstArr,
                  int E, int N,
                  const unsigned* __restrict__ bp1, const float* __restrict__ b1,
                  const unsigned* __restrict__ bp2, const float* __restrict__ b2,
                  unsigned* __restrict__ enc)
{
  constexpr int DIN  = CF + 3;
  constexpr int DINP = ((DIN + 31) / 32) * 32;
  constexpr int NT   = DOUT / 16;
  constexpr int K1   = DINP / 32;
  constexpr int K2   = DOUT / 32;
  constexpr int MB   = 32;                 // edges per workgroup (2 M-tiles)

  __shared__ unsigned short sMsg[MB][DINP];
  __shared__ unsigned short sH1[MB][DOUT];
  __shared__ int sSrc[MB], sDst[MB];

  const int tid  = threadIdx.x;
  const int lane = tid & 31;
  const int wave = tid >> 5;
  const int hf   = lane >> 4;
  const int ln16 = lane & 15;
  const long eBase = (long)blockIdx.x * MB;
  const long ET = (long)E + N;

  // ---- edge indices (self-loops appended after the E real edges) ----
  if (tid < MB) {
    long e = eBase + tid;
    int s = -1, d = -1;
    if (e < E)       { s = srcArr[e]; d = dstArr[e]; }
    else if (e < ET) { s = d = (int)(e - E); }
    sSrc[tid] = s; sDst[tid] = d;
  }
  __syncthreads();

  // ---- gather message tile into LDS (bf16, zero-padded K) ----
  for (int i = tid; i < MB * DINP; i += 256) {
    int m = i / DINP, j = i - m * DINP;
    int s = sSrc[m], d = sDst[m];
    float val = 0.f;
    if (s >= 0) {
      if (j < CF)       val = feat[(long)s * CF + j];
      else if (j < DIN) val = pos[(long)s * 3 + (j - CF)] - pos[(long)d * 3 + (j - CF)];
    }
    sMsg[m][j] = f2bf(val);
  }
  __syncthreads();

  // ---- GEMM1: H1 = relu(msg @ W1 + b1) ----
  for (int item = wave; item < 2 * NT; item += 8) {
    const int mt = item / NT, nt = item % NT;
    v8f acc = {};
    for (int ks = 0; ks < K1; ++ks) {
      ABu a;
#pragma unroll
      for (int v = 0; v < 8; ++v) {
        int kb = ks * 32 + 2 * v + 16 * hf;
        a.u[v] = *(const unsigned*)&sMsg[mt * 16 + ln16][kb];
      }
      const unsigned* bq = bp1 + ((long)(ks * NT + nt) * 256) + lane * 8;
      __builtin_prefetch(bq + (long)NT * 256, 0, 0);     // -> global_prefetch_b8
      ABu b;
      *(uint4*)&b.u[0] = *(const uint4*)bq;
      *(uint4*)&b.u[4] = *(const uint4*)(bq + 4);
      acc = __builtin_amdgcn_wmma_f32_16x16x32_bf16(
                false, a.v, false, b.v, (short)0, acc, false, false);
    }
    const float bias = b1[nt * 16 + ln16];
#pragma unroll
    for (int r = 0; r < 8; ++r) {
      float h = acc[r] + bias;
      h = h > 0.f ? h : 0.f;
      sH1[mt * 16 + r + 8 * hf][nt * 16 + ln16] = f2bf(h);
    }
  }
  __syncthreads();

  // ---- GEMM2: out = H1 @ W2 + b2 ; scatter-max to enc[dst] ----
  for (int item = wave; item < 2 * NT; item += 8) {
    const int mt = item / NT, nt = item % NT;
    v8f acc = {};
    for (int ks = 0; ks < K2; ++ks) {
      ABu a;
#pragma unroll
      for (int v = 0; v < 8; ++v) {
        int kb = ks * 32 + 2 * v + 16 * hf;
        a.u[v] = *(const unsigned*)&sH1[mt * 16 + ln16][kb];
      }
      const unsigned* bq = bp2 + ((long)(ks * NT + nt) * 256) + lane * 8;
      __builtin_prefetch(bq + (long)NT * 256, 0, 0);
      ABu b;
      *(uint4*)&b.u[0] = *(const uint4*)bq;
      *(uint4*)&b.u[4] = *(const uint4*)(bq + 4);
      acc = __builtin_amdgcn_wmma_f32_16x16x32_bf16(
                false, a.v, false, b.v, (short)0, acc, false, false);
    }
    const float bias = b2[nt * 16 + ln16];
    const int n = nt * 16 + ln16;
#pragma unroll
    for (int r = 0; r < 8; ++r) {
      int d = sDst[mt * 16 + r + 8 * hf];
      if (d >= 0) {
        unsigned ev = encf(acc[r] + bias);
        atomicMax(&enc[(long)d * DOUT + n], ev);     // global_atomic_max_u32
      }
    }
  }
}

// ---------------------------------------------------------------------------
// projection + scatter-add into (256,1216,240) image; 1 block = 1 point
// ---------------------------------------------------------------------------
__global__ __launch_bounds__(256)
void scatter_k(const float* __restrict__ feat3, const float* __restrict__ pos,
               const float* __restrict__ proj, const int* __restrict__ imgH,
               const int* __restrict__ imgW, float* __restrict__ img, int N) {
  const int i = blockIdx.x;
  if (i >= N) return;
  __shared__ int sui, svi, sval;
  if (threadIdx.x == 0) {
    float px = pos[3 * i], py = pos[3 * i + 1], pz = pos[3 * i + 2];
    float h0 = proj[0] * px + proj[1] * py + proj[2]  * pz + proj[3];
    float h1 = proj[4] * px + proj[5] * py + proj[6]  * pz + proj[7];
    float h2 = proj[8] * px + proj[9] * py + proj[10] * pz + proj[11];
    int offu = imgW[0] - 1216, offv = imgH[0] - 240;
    int ui = (int)(short)(h0 / h2) - offu;
    int vi = (int)(short)(h1 / h2) - offv;
    sval = (ui >= 0) & (vi >= 0) & (ui < 1216) & (vi < 240);
    sui = ui; svi = vi;
  }
  __syncthreads();
  if (sval) {
    int c = threadIdx.x;                               // 256 channels
    atomicAdd(&img[(long)c * (1216 * 240) + sui * 240 + svi],
              feat3[(long)i * 256 + c]);               // global_atomic_add_f32
  }
}

// 4x4 max-pool (256,1216,240) -> (256,304,60)
__global__ void pool_k(const float* __restrict__ img, float* __restrict__ out) {
  const long total = 256L * 304 * 60;
  long idx = (long)blockIdx.x * 256 + threadIdx.x;
  if (idx >= total) return;
  int ho = (int)(idx % 60);
  long t = idx / 60;
  int wo = (int)(t % 304);
  int c  = (int)(t / 304);
  const float* base = img + (long)c * (1216 * 240) + (long)(wo * 4) * 240 + ho * 4;
  float mx = -3.402823466e38f;
#pragma unroll
  for (int a = 0; a < 4; ++a)
#pragma unroll
    for (int q = 0; q < 4; ++q) {
      float v = base[a * 240 + q];
      mx = v > mx ? v : mx;
    }
  out[idx] = mx;
}

// ---------------------------------------------------------------------------
extern "C" void kernel_launch(void* const* d_in, const int* in_sizes, int n_in,
                              void* d_out, int out_size, void* d_ws, size_t ws_size,
                              hipStream_t stream) {
  const float* x    = (const float*)d_in[0];
  const float* pos  = (const float*)d_in[1];
  const int*   ei   = (const int*)  d_in[2];
  const float* proj = (const float*)d_in[4];
  const int*   imgH = (const int*)  d_in[5];
  const int*   imgW = (const int*)  d_in[6];
  const float* w1_1 = (const float*)d_in[7];
  const float* b1_1 = (const float*)d_in[8];
  const float* w2_1 = (const float*)d_in[9];
  const float* b2_1 = (const float*)d_in[10];
  const float* w1_2 = (const float*)d_in[11];
  const float* b1_2 = (const float*)d_in[12];
  const float* w2_2 = (const float*)d_in[13];
  const float* b2_2 = (const float*)d_in[14];
  const float* w1_3 = (const float*)d_in[15];
  const float* b1_3 = (const float*)d_in[16];
  const float* w2_3 = (const float*)d_in[17];
  const float* b2_3 = (const float*)d_in[18];

  const int N = in_sizes[0] / 4;
  const int E = in_sizes[2] / 2;
  const int* srcArr = ei;
  const int* dstArr = ei + E;

  // ---- workspace carve-up ----
  char* ws = (char*)d_ws;
  size_t off = 0;
  auto carve = [&](size_t bytes) -> void* {
    void* p = ws + off;
    off += (bytes + 255) & ~(size_t)255;
    return p;
  };
  unsigned* enc1 = (unsigned*)carve((size_t)N * 64  * 4);
  unsigned* enc2 = (unsigned*)carve((size_t)N * 128 * 4);
  unsigned* enc3 = (unsigned*)carve((size_t)N * 256 * 4);
  float*    img  = (float*)   carve(256L * 1216 * 240 * 4);
  unsigned* bp1a = (unsigned*)carve(32L  * 64  * 2);   // W1_1 packed (Kp=32)
  unsigned* bp1b = (unsigned*)carve(64L  * 64  * 2);   // W2_1 packed
  unsigned* bp2a = (unsigned*)carve(96L  * 128 * 2);   // W1_2 packed (Kp=96)
  unsigned* bp2b = (unsigned*)carve(128L * 128 * 2);   // W2_2 packed
  unsigned* bp3a = (unsigned*)carve(160L * 256 * 2);   // W1_3 packed (Kp=160)
  unsigned* bp3b = (unsigned*)carve(256L * 256 * 2);   // W2_3 packed
  (void)ws_size; (void)n_in; (void)out_size;

  // ---- zero max-buffers (0 == encoded -inf) and the scatter image ----
  hipMemsetAsync(enc1, 0, (size_t)N * 64  * 4, stream);
  hipMemsetAsync(enc2, 0, (size_t)N * 128 * 4, stream);
  hipMemsetAsync(enc3, 0, (size_t)N * 256 * 4, stream);
  hipMemsetAsync(img,  0, 256L * 1216 * 240 * 4, stream);

  // ---- weight packing (tiny) ----
  auto packLaunch = [&](const float* w, unsigned* bp, int Kreal, int Kp, int Nout) {
    int total = (Kp / 32) * (Nout / 16) * 256;
    pack_w<<<(total + 255) / 256, 256, 0, stream>>>(w, bp, Kreal, Kp, Nout);
  };
  packLaunch(w1_1, bp1a,   7,  32,  64);
  packLaunch(w2_1, bp1b,  64,  64,  64);
  packLaunch(w1_2, bp2a,  67,  96, 128);
  packLaunch(w2_2, bp2b, 128, 128, 128);
  packLaunch(w1_3, bp3a, 131, 160, 256);
  packLaunch(w2_3, bp3b, 256, 256, 256);

  const long ET = (long)E + N;
  const int  nblk = (int)((ET + 31) / 32);

  // ---- layer 1: (x[4], pos) -> 64 ----
  layer_kernel<4, 64><<<nblk, 256, 0, stream>>>(
      x, pos, srcArr, dstArr, E, N, bp1a, b1_1, bp1b, b2_1, enc1);
  decode_k<<<(int)(((long)N * 64 + 255) / 256), 256, 0, stream>>>(enc1, (long)N * 64);

  // ---- layer 2: 64 -> 128 ----
  layer_kernel<64, 128><<<nblk, 256, 0, stream>>>(
      (const float*)enc1, pos, srcArr, dstArr, E, N, bp2a, b1_2, bp2b, b2_2, enc2);
  decode_k<<<(int)(((long)N * 128 + 255) / 256), 256, 0, stream>>>(enc2, (long)N * 128);

  // ---- layer 3: 128 -> 256 ----
  layer_kernel<128, 256><<<nblk, 256, 0, stream>>>(
      (const float*)enc2, pos, srcArr, dstArr, E, N, bp3a, b1_3, bp3b, b2_3, enc3);
  decode_k<<<(int)(((long)N * 256 + 255) / 256), 256, 0, stream>>>(enc3, (long)N * 256);

  // ---- projection + scatter-add, then 4x4 max-pool into d_out ----
  scatter_k<<<N, 256, 0, stream>>>((const float*)enc3, pos, proj, imgH, imgW, img, N);

  const long ptot = 256L * 304 * 60;
  pool_k<<<(int)((ptot + 255) / 256), 256, 0, stream>>>(img, (float*)d_out);
}
